// AttnMatch_10754598109877
// MI455X (gfx1250) — matrix operand order repaired and checked by reading, not compiled, not run on hardware
//
#include <hip/hip_runtime.h>

#define DK 1024   // D
#define SS 2048   // S
#define BB 32     // B

typedef __attribute__((ext_vector_type(16))) _Float16 v16h;
typedef __attribute__((ext_vector_type(8)))  float    v8f;

union HFrag { v16h v; uint4 q[2]; };

__device__ __forceinline__ unsigned int pk2h(float a, float b) {
    union { _Float16 h[2]; unsigned int u; } x;
    x.h[0] = (_Float16)a; x.h[1] = (_Float16)b;
    return x.u;
}

// ---------------------------------------------------------------------------
// Kernel 0: one-shot W f32 -> f16 conversion (4 MB read, 2 MB write)
// ---------------------------------------------------------------------------
__global__ void wcvt_kernel(const float* __restrict__ W,
                            _Float16* __restrict__ Wh) {
    const int i = (blockIdx.x * 256 + threadIdx.x) * 4;   // DK*DK elements
    float4 f = *(const float4*)(W + i);
    uint2 p; p.x = pk2h(f.x, f.y); p.y = pk2h(f.z, f.w);
    *(uint2*)(Wh + i) = p;
}

// ---------------------------------------------------------------------------
// Kernel 1: bias[b,e] = sum_d rs[b,d] * U[e,d]   (tiny GEMV, one wave per e)
// ---------------------------------------------------------------------------
__global__ void bias_kernel(const float* __restrict__ rs,
                            const float* __restrict__ U,
                            float* __restrict__ bias) {
    const int blk  = blockIdx.x;          // BB * (DK/8)
    const int b    = blk >> 7;
    const int eg   = blk & 127;
    const int wave = threadIdx.x >> 5;
    const int lane = threadIdx.x & 31;
    const int e    = eg * 8 + wave;
    const float* urow = U  + (size_t)e * DK;
    const float* rrow = rs + (size_t)b * DK;
    float acc = 0.f;
    for (int d = lane; d < DK; d += 32) acc += urow[d] * rrow[d];
    #pragma unroll
    for (int off = 16; off > 0; off >>= 1) acc += __shfl_xor(acc, off, 32);
    if (lane == 0) bias[b * DK + e] = acc;
}

// ---------------------------------------------------------------------------
// Kernel 2: fused  scores[b,s] = sum_e relu( (A W^T)[s,e] + bias[b,e] ) * rs[b,e]
// A strip (128 x 1024) staged f32->f16 in LDS (padded rows).
// Each wave owns one 16-row m-tile; per pass it computes TWO 16x16 C tiles
// (independent accumulators c0/c1 sharing one A fragment) -> 2 WMMA per DS read.
// ---------------------------------------------------------------------------
__global__ void __launch_bounds__(256, 1)
attn_score_kernel(const float* __restrict__ att,      // [B,S,D] f32
                  const float* __restrict__ rs,       // [B,D]   f32
                  const _Float16* __restrict__ Wh,    // [D,D]   f16, row = e
                  const float* __restrict__ bias,     // [B,D]   f32
                  float* __restrict__ scores) {       // [B,S]   f32
    extern __shared__ __align__(16) char smem[];
    const int tid   = threadIdx.x;
    const int lane  = tid & 31;
    const int mtile = tid >> 5;               // wave id 0..7 -> m-tile
    const int bid   = blockIdx.x;             // BB * 16
    const int b     = bid >> 4;
    const int s0    = (bid & 15) * 128;

    const int ROWB = (DK + 8) * 2;            // 2064 B per padded LDS row

    // ---- stage A strip: f32 -> f16, 8B (4 halves) per thread-iteration
    const float* abase = att + ((size_t)b * SS + s0) * DK;
    for (int i = tid; i < (128 * DK) / 4; i += 256) {
        const int row = i >> 8;               // 256 float4 per row
        const int c4  = i & 255;
        float4 f = ((const float4*)(abase + (size_t)row * DK))[c4];
        uint2 p; p.x = pk2h(f.x, f.y); p.y = pk2h(f.z, f.w);
        *(uint2*)(smem + row * ROWB + c4 * 8) = p;
    }
    __syncthreads();

    const int hi16 = lane >> 4;               // lane half selects K sub-block
    const char* arow = smem + (mtile * 16 + (lane & 15)) * ROWB;

    float sacc[8];
    #pragma unroll
    for (int v = 0; v < 8; ++v) sacc[v] = 0.f;

    for (int npass = 0; npass < DK / 32; ++npass) {       // 32 cols per pass
        const int ncol0 = npass * 32 + (lane & 15);       // W rows (= out cols)
        const int ncol1 = ncol0 + 16;
        const _Float16* w0 = Wh + (size_t)ncol0 * DK + hi16 * 16;
        const _Float16* w1 = Wh + (size_t)ncol1 * DK + hi16 * 16;

        v8f c0 = {}, c1 = {};
        #pragma unroll 2
        for (int kb = 0; kb < DK; kb += 32) {
            // A fragment (contiguous K: lanes 0-15 K 0-15; lanes 16-31 K 16-31)
            HFrag a;
            a.q[0] = *(const uint4*)(arow + (kb + hi16 * 16) * 2);
            a.q[1] = *(const uint4*)(arow + (kb + hi16 * 16 + 8) * 2);
            // B fragments: 16 contiguous f16 K-values of columns ncol0 / ncol1
            HFrag b0, b1;
            b0.q[0] = *(const uint4*)(w0 + kb);
            b0.q[1] = *(const uint4*)(w0 + kb + 8);
            b1.q[0] = *(const uint4*)(w1 + kb);
            b1.q[1] = *(const uint4*)(w1 + kb + 8);
            c0 = __builtin_amdgcn_wmma_f32_16x16x32_f16(
                     false, a.v, false, b0.v, (short)0, c0, false, false);
            c1 = __builtin_amdgcn_wmma_f32_16x16x32_f16(
                     false, a.v, false, b1.v, (short)0, c1, false, false);
        }
        // epilogue: bias + relu + dot with rs, accumulate per-row partial score
        const float bv0 = bias[b * DK + ncol0];
        const float rv0 = rs[b * DK + ncol0];
        const float bv1 = bias[b * DK + ncol1];
        const float rv1 = rs[b * DK + ncol1];
        #pragma unroll
        for (int v = 0; v < 8; ++v) {
            float t0 = c0[v] + bv0; t0 = t0 > 0.f ? t0 : 0.f;
            float t1 = c1[v] + bv1; t1 = t1 > 0.f ? t1 : 0.f;
            sacc[v] += t0 * rv0 + t1 * rv1;
        }
    }

    // reduce across the 16 lanes of each half (columns) -> score per row
    #pragma unroll
    for (int v = 0; v < 8; ++v) {
        float a = sacc[v];
        a += __shfl_xor(a, 1, 32);
        a += __shfl_xor(a, 2, 32);
        a += __shfl_xor(a, 4, 32);
        a += __shfl_xor(a, 8, 32);
        sacc[v] = a;
    }
    if ((lane & 15) == 0) {
        const int mbase = mtile * 16 + hi16 * 8;   // C layout: lanes>=16 -> M=8..15
        #pragma unroll
        for (int v = 0; v < 8; ++v)
            scores[b * SS + s0 + mbase + v] = sacc[v];
    }
}

// ---------------------------------------------------------------------------
// Kernel 3: in-place softmax over S per batch row
// ---------------------------------------------------------------------------
__global__ void softmax_kernel(float* __restrict__ scores) {
    __shared__ float red[256];
    const int b   = blockIdx.x;
    const int tid = threadIdx.x;
    float* row = scores + (size_t)b * SS;
    float v[8];
    float m = -3.0e38f;
    #pragma unroll
    for (int i = 0; i < 8; ++i) { v[i] = row[tid + i * 256]; m = fmaxf(m, v[i]); }
    red[tid] = m; __syncthreads();
    for (int off = 128; off > 0; off >>= 1) {
        if (tid < off) red[tid] = fmaxf(red[tid], red[tid + off]);
        __syncthreads();
    }
    m = red[0]; __syncthreads();
    float s = 0.f;
    #pragma unroll
    for (int i = 0; i < 8; ++i) { v[i] = __expf(v[i] - m); s += v[i]; }
    red[tid] = s; __syncthreads();
    for (int off = 128; off > 0; off >>= 1) {
        if (tid < off) red[tid] += red[tid + off];
        __syncthreads();
    }
    const float inv = 1.0f / red[0];
    #pragma unroll
    for (int i = 0; i < 8; ++i) row[tid + i * 256] = v[i] * inv;
}

// ---------------------------------------------------------------------------
// Kernel 4: split-S partial weighted sums (deterministic, no float atomics)
// ---------------------------------------------------------------------------
__global__ void matched_partial_kernel(const float* __restrict__ att,
                                       const float* __restrict__ wgt,
                                       float* __restrict__ part) {
    const int d  = blockIdx.x * 256 + threadIdx.x;   // [0, DK)
    const int sc = blockIdx.y;                       // [0, 8)
    const int b  = blockIdx.z;                       // [0, BB)
    const float* ab = att + ((size_t)b * SS + sc * 256) * DK + d;
    const float* wb = wgt + (size_t)b * SS + sc * 256;
    float acc = 0.f;
    for (int s = 0; s < 256; ++s) acc += ab[(size_t)s * DK] * wb[s];
    part[((size_t)b * 8 + sc) * DK + d] = acc;
}

// ---------------------------------------------------------------------------
// Kernel 5: fixed-order reduction of the 8 partials -> matched_seq[b,d]
// ---------------------------------------------------------------------------
__global__ void matched_reduce_kernel(const float* __restrict__ part,
                                      float* __restrict__ matched) {
    const int idx = blockIdx.x * 256 + threadIdx.x;  // [0, BB*DK)
    const int b = idx >> 10;
    const int d = idx & 1023;
    float acc = 0.f;
    #pragma unroll
    for (int sc = 0; sc < 8; ++sc) acc += part[((size_t)b * 8 + sc) * DK + d];
    matched[idx] = acc;
}

// ---------------------------------------------------------------------------
extern "C" void kernel_launch(void* const* d_in, const int* in_sizes, int n_in,
                              void* d_out, int out_size, void* d_ws, size_t ws_size,
                              hipStream_t stream) {
    const float* att = (const float*)d_in[0];   // [B,S,D]
    const float* rs  = (const float*)d_in[1];   // [1,B,D] -> [B,D]
    const float* W   = (const float*)d_in[2];   // [D,D]
    const float* U   = (const float*)d_in[3];   // [D,D]

    float* matched = (float*)d_out;             // [B,D]   (output 0)
    float* weights = (float*)d_out + BB * DK;   // [B,S]   (output 1; scores in-place)

    // workspace layout
    float*     bias = (float*)d_ws;                       // [B,D]    128 KB
    float*     part = bias + BB * DK;                     // [B,8,D]  1 MB
    _Float16*  Wh   = (_Float16*)(part + BB * 8 * DK);    // [D,D]    2 MB

    wcvt_kernel<<<dim3(DK * DK / 1024), dim3(256), 0, stream>>>(W, Wh);

    bias_kernel<<<dim3(BB * (DK / 8)), dim3(256), 0, stream>>>(rs, U, bias);

    const size_t smem = 128 * (size_t)((DK + 8) * 2);     // 258 KB (<= 320 KB/WGP)
    attn_score_kernel<<<dim3(BB * (SS / 128)), dim3(256), smem, stream>>>(
        att, rs, Wh, bias, weights);

    softmax_kernel<<<dim3(BB), dim3(256), 0, stream>>>(weights);

    matched_partial_kernel<<<dim3(DK / 256, 8, BB), dim3(256), 0, stream>>>(
        att, weights, part);

    matched_reduce_kernel<<<dim3(BB * DK / 256), dim3(256), 0, stream>>>(
        part, matched);
}
